// ProductAttention_20194936225916
// MI455X (gfx1250) — matrix operand order repaired and checked by reading, not compiled, not run on hardware
//
#include <hip/hip_runtime.h>

typedef __attribute__((ext_vector_type(16))) _Float16 v16h;
typedef __attribute__((ext_vector_type(8)))  _Float16 v8h;
typedef __attribute__((ext_vector_type(8)))  float    v8f;

constexpr int B_ = 2, C_ = 128, H_ = 128, W_ = 128;
constexpr int HW_ = H_ * W_;
constexpr int NH_ = 4, HD_ = 32, K_ = 16, OFF_ = 4;

__device__ __forceinline__ int reflect(int i, int L) {
  if (i < 0) i = -i;
  if (i >= L) i = 2 * (L - 1) - i;
  return i;
}

// ---------------------------------------------------------------------------
// f32 -> f16 weight conversion (pointwise weight matrices, 128x128 each)
// ---------------------------------------------------------------------------
__global__ void cvt_f16(const float* __restrict__ s, _Float16* __restrict__ d, int n) {
  int i = blockIdx.x * 256 + threadIdx.x;
  if (i < n) d[i] = (_Float16)s[i];
}

// ---------------------------------------------------------------------------
// Fused depthwise 3x3 (zero pad) + bias + ReLU for q,k,v. Reads vid once.
// Output layout: pixel-major [b][hw][c] in f16 (feeds WMMA B-fragments).
// ---------------------------------------------------------------------------
__global__ __launch_bounds__(128) void dw3x3_relu(
    const float* __restrict__ x,
    const float* __restrict__ qw, const float* __restrict__ qb,
    const float* __restrict__ kw, const float* __restrict__ kb,
    const float* __restrict__ vw, const float* __restrict__ vb,
    _Float16* __restrict__ yq, _Float16* __restrict__ yk, _Float16* __restrict__ yv)
{
  int w = threadIdx.x, h = blockIdx.x, c = blockIdx.y, b = blockIdx.z;
  const float* xp = x + ((size_t)b * C_ + c) * (size_t)HW_;
  float sq = 0.f, sk = 0.f, sv = 0.f;
#pragma unroll
  for (int dy = -1; dy <= 1; ++dy) {
    int hh = h + dy;
    bool hv = (unsigned)hh < (unsigned)H_;
#pragma unroll
    for (int dx = -1; dx <= 1; ++dx) {
      int ww = w + dx;
      float xv = (hv && (unsigned)ww < (unsigned)W_) ? xp[hh * W_ + ww] : 0.f;
      int wi9 = c * 9 + (dy + 1) * 3 + (dx + 1);
      sq += xv * qw[wi9];
      sk += xv * kw[wi9];
      sv += xv * vw[wi9];
    }
  }
  sq = fmaxf(sq + qb[c], 0.f);
  sk = fmaxf(sk + kb[c], 0.f);
  sv = fmaxf(sv + vb[c], 0.f);
  size_t o = ((size_t)b * HW_ + h * W_ + w) * C_ + c;
  yq[o] = (_Float16)sq;
  yk[o] = (_Float16)sk;
  yv[o] = (_Float16)sv;
}

// ---------------------------------------------------------------------------
// WMMA GEMM, register-blocked over N:
//   D[b][n][m] = act( W[m][k] * Y[b][n][k] + bias[m] ) * scale
// M=K=128, N=HW per batch. Block: 256 thr = 8 waves; wave = one 16-row M tile
// x four 16-pixel N sub-tiles (64 pixels/block). A fragment loaded once per
// k-step and reused across the 4 sub-tiles -> 16 v_wmma per wave.
// MODE 0: f16 out, pixel-major [b][n][m].  MODE 1: f32 out, [b][m][hw] (d_out).
// ---------------------------------------------------------------------------
template <int MODE>
__global__ __launch_bounds__(256) void pw_gemm(
    const _Float16* __restrict__ Wt,   // [128][128] f16, row = out-ch, col = in-ch
    const float* __restrict__ bias,    // [128]
    const _Float16* __restrict__ Y,    // [B][HW][128] f16
    void* __restrict__ D, float scale)
{
  int lane = threadIdx.x & 31;
  int wav  = threadIdx.x >> 5;
  int l16  = lane & 15;
  int half = lane >> 4;
  int mbase = wav * 16;
  int n0 = blockIdx.x * 64;          // 4 N sub-tiles of 16
  int b  = blockIdx.y;

  const _Float16* ybase = Y + ((size_t)b * HW_ + n0 + l16) * C_;  // K contiguous
  const _Float16* arow  = Wt + (size_t)(mbase + l16) * C_;
  int klo = half * 16;

  v8f acc[4] = {};
#pragma unroll
  for (int ks = 0; ks < C_; ks += 32) {
    // A fragment: 16 consecutive K values klo..klo+15 per lane (matches B)
    v8h alo = *(const v8h*)(arow + ks + klo);
    v8h ahi = *(const v8h*)(arow + ks + klo + 8);
    v16h a;
#pragma unroll
    for (int i = 0; i < 8; ++i) { a[i] = alo[i]; a[8 + i] = ahi[i]; }
#pragma unroll
    for (int sub = 0; sub < 4; ++sub) {
      // B fragment: 16 consecutive K values per lane (same contiguous order as A)
      v16h bf = *(const v16h*)(ybase + (size_t)sub * 16 * C_ + ks + half * 16);
      acc[sub] = __builtin_amdgcn_wmma_f32_16x16x32_f16(false, a, false, bf,
                                                        (short)0, acc[sub], false, false);
    }
  }

  int mo = mbase + half * 8;                 // C/D layout: VGPR r -> M = mo + r
  float bs[8];
#pragma unroll
  for (int r = 0; r < 8; ++r) bs[r] = bias[mo + r];

#pragma unroll
  for (int sub = 0; sub < 4; ++sub) {
    int n = n0 + sub * 16 + l16;
    if (MODE == 0) {
      _Float16* drow = (_Float16*)D + ((size_t)b * HW_ + n) * C_ + mo;
      v8h o;
#pragma unroll
      for (int r = 0; r < 8; ++r) o[r] = (_Float16)((acc[sub][r] + bs[r]) * scale);
      *(v8h*)drow = o;
    } else {
      float* dp = (float*)D + ((size_t)b * C_ + mo) * (size_t)HW_ + n;
#pragma unroll
      for (int r = 0; r < 8; ++r) dp[(size_t)r * HW_] = acc[sub][r] + bs[r];
    }
  }
}

// ---------------------------------------------------------------------------
// Neighborhood attention: one wave32 per (b, head, pixel).
// Phase 1: 2 of 64 shifted dot-products per lane (q broadcast via LDS).
// Phase 2: top-16 via 16 wave-wide argmax butterflies (__shfl_xor).
// Phase 3: softmax (replicated per lane). Phase 4: lane = head-dim channel,
// gather v (32 contiguous f16 across lanes) and weighted-sum.
// ---------------------------------------------------------------------------
__global__ __launch_bounds__(256) void attn_topk(
    const _Float16* __restrict__ q, const _Float16* __restrict__ k,
    const _Float16* __restrict__ v, _Float16* __restrict__ o)
{
  __shared__ float sq[8][HD_];
  int lane = threadIdx.x & 31, wav = threadIdx.x >> 5;
  unsigned gw = blockIdx.x * 8u + wav;       // (b*NH + nh)*HW + p
  int p  = gw & (HW_ - 1);
  int nh = (gw >> 14) & (NH_ - 1);
  int b  = gw >> 16;
  int h = p >> 7, w = p & (W_ - 1);
  size_t rb = (size_t)b * HW_;

  const _Float16* qr = q + (rb + p) * C_ + nh * HD_;
  sq[wav][lane] = (float)qr[lane];
  __syncthreads();

  float d0 = 0.f, d1 = 0.f;
#pragma unroll
  for (int t = 0; t < 2; ++t) {
    int off = 2 * lane + t;
    int h2 = reflect(h + (off >> 3) - OFF_, H_);
    int w2 = reflect(w + (off & 7) - OFF_, W_);
    const _Float16* kr = k + (rb + h2 * W_ + w2) * C_ + nh * HD_;
    float acc = 0.f;
#pragma unroll
    for (int c0 = 0; c0 < HD_; c0 += 8) {
      v8h kk = *(const v8h*)(kr + c0);
#pragma unroll
      for (int j = 0; j < 8; ++j) acc += sq[wav][c0 + j] * (float)kk[j];
    }
    if (t == 0) d0 = acc; else d1 = acc;
  }

  const float NEG = -3.402823466e38f;
  float wv[K_]; int wi[K_];
#pragma unroll
  for (int i = 0; i < K_; ++i) {
    float bv; int bi;
    if (d0 >= d1) { bv = d0; bi = 2 * lane; } else { bv = d1; bi = 2 * lane + 1; }
#pragma unroll
    for (int s = 16; s > 0; s >>= 1) {   // wave32 butterfly argmax (tie: low idx)
      float ov = __shfl_xor(bv, s, 32);
      int   oi = __shfl_xor(bi, s, 32);
      if (ov > bv || (ov == bv && oi < bi)) { bv = ov; bi = oi; }
    }
    wv[i] = bv; wi[i] = bi;
    if (bi == 2 * lane) d0 = NEG;
    else if (bi == 2 * lane + 1) d1 = NEG;
  }

  float m = wv[0], s = 0.f, e[K_];       // extraction order is descending
#pragma unroll
  for (int i = 0; i < K_; ++i) { e[i] = __expf(wv[i] - m); s += e[i]; }
  float inv = 1.f / s;

  float out = 0.f;
#pragma unroll
  for (int i = 0; i < K_; ++i) {
    int h2 = reflect(h + (wi[i] >> 3) - OFF_, H_);
    int w2 = reflect(w + (wi[i] & 7) - OFF_, W_);
    out += e[i] * inv * (float)v[(rb + h2 * W_ + w2) * C_ + nh * HD_ + lane];
  }
  o[(rb + p) * C_ + nh * HD_ + lane] = (_Float16)out;
}

// ---------------------------------------------------------------------------
extern "C" void kernel_launch(void* const* d_in, const int* in_sizes, int n_in,
                              void* d_out, int out_size, void* d_ws, size_t ws_size,
                              hipStream_t stream) {
  (void)in_sizes; (void)n_in; (void)out_size; (void)ws_size;
  const float* vid    = (const float*)d_in[0];
  const float* q_dw_w = (const float*)d_in[1];
  const float* q_dw_b = (const float*)d_in[2];
  const float* q_pw_w = (const float*)d_in[3];
  const float* q_pw_b = (const float*)d_in[4];
  const float* k_dw_w = (const float*)d_in[5];
  const float* k_dw_b = (const float*)d_in[6];
  const float* k_pw_w = (const float*)d_in[7];
  const float* k_pw_b = (const float*)d_in[8];
  const float* v_dw_w = (const float*)d_in[9];
  const float* v_dw_b = (const float*)d_in[10];
  const float* v_pw_w = (const float*)d_in[11];
  const float* v_pw_b = (const float*)d_in[12];
  const float* proj_w = (const float*)d_in[13];
  const float* proj_b = (const float*)d_in[14];

  char* ws = (char*)d_ws;
  size_t off = 0;
  auto take = [&](size_t bytes) -> char* {
    char* p = ws + off;
    off += (bytes + 255) & ~(size_t)255;
    return p;
  };
  const size_t actB = (size_t)B_ * HW_ * C_ * sizeof(_Float16);  // 8 MB each
  _Float16* wq16 = (_Float16*)take(C_ * C_ * sizeof(_Float16));
  _Float16* wk16 = (_Float16*)take(C_ * C_ * sizeof(_Float16));
  _Float16* wv16 = (_Float16*)take(C_ * C_ * sizeof(_Float16));
  _Float16* wp16 = (_Float16*)take(C_ * C_ * sizeof(_Float16));
  _Float16* yq   = (_Float16*)take(actB);
  _Float16* yk   = (_Float16*)take(actB);
  _Float16* yv   = (_Float16*)take(actB);
  _Float16* qb   = (_Float16*)take(actB);
  _Float16* kb   = (_Float16*)take(actB);
  _Float16* vb   = (_Float16*)take(actB);
  _Float16* ab   = yq;  // dead after q GEMM; reuse for attention output

  // 1. weights -> f16
  cvt_f16<<<64, 256, 0, stream>>>(q_pw_w, wq16, C_ * C_);
  cvt_f16<<<64, 256, 0, stream>>>(k_pw_w, wk16, C_ * C_);
  cvt_f16<<<64, 256, 0, stream>>>(v_pw_w, wv16, C_ * C_);
  cvt_f16<<<64, 256, 0, stream>>>(proj_w, wp16, C_ * C_);

  // 2. fused depthwise + ReLU for q,k,v (reads vid once)
  dw3x3_relu<<<dim3(H_, C_, B_), 128, 0, stream>>>(
      vid, q_dw_w, q_dw_b, k_dw_w, k_dw_b, v_dw_w, v_dw_b, yq, yk, yv);

  // 3. pointwise GEMMs via WMMA (q folds in scale = HD^-0.5)
  const float qscale = 0.17677669529663687f;  // 32^-0.5
  dim3 ggrid(HW_ / 64, B_);
  pw_gemm<0><<<ggrid, 256, 0, stream>>>(wq16, q_pw_b, yq, qb, qscale);
  pw_gemm<0><<<ggrid, 256, 0, stream>>>(wk16, k_pw_b, yk, kb, 1.0f);
  pw_gemm<0><<<ggrid, 256, 0, stream>>>(wv16, v_pw_b, yv, vb, 1.0f);

  // 4. neighborhood top-k attention: one wave per (b, head, pixel)
  attn_topk<<<(B_ * NH_ * HW_) / 8, 256, 0, stream>>>(qb, kb, vb, ab);

  // 5. output projection via WMMA, f32 channel-major straight into d_out
  pw_gemm<1><<<ggrid, 256, 0, stream>>>(wp16, proj_b, ab, d_out, 1.0f);
}
